// LocalAttention_55825984913525
// MI455X (gfx1250) — matrix operand order repaired and checked by reading, not compiled, not run on hardware
//
#include <hip/hip_runtime.h>

typedef unsigned short ushort_t;
typedef unsigned int uint_t;
typedef __attribute__((ext_vector_type(16))) __bf16 v16bf;
typedef __attribute__((ext_vector_type(8)))  float  v8f;
typedef __attribute__((ext_vector_type(4)))  unsigned int u32x4;
typedef __attribute__((ext_vector_type(4)))  int v4i;

#define LDSW 40   // padded LDS row stride in bf16 elements (80 B, 16B-aligned)

#if defined(__has_builtin)
#if __has_builtin(__builtin_amdgcn_global_load_async_to_lds_b128) && \
    __has_builtin(__builtin_amdgcn_s_wait_asynccnt)
#define USE_ASYNC_LDS 1
#endif
#endif

#ifdef USE_ASYNC_LDS
typedef __attribute__((address_space(1))) v4i* gv4i_t;   // global <4 x i32>*
typedef __attribute__((address_space(3))) v4i* lv4i_t;   // LDS    <4 x i32>*
#endif

// ---------- bf16 helpers (raw-bit, RNE) ----------
__device__ __forceinline__ ushort_t f2bf(float f) {
    uint_t u = __float_as_uint(f);
    u = (u + 0x7FFFu + ((u >> 16) & 1u)) >> 16;
    return (ushort_t)u;
}
__device__ __forceinline__ float bf2f(ushort_t h) {
    return __uint_as_float(((uint_t)h) << 16);
}

// ---------- fp32 -> bf16 conversion (grid-stride) ----------
__global__ void cvt_f32_bf16_kernel(const float* __restrict__ in,
                                    ushort_t* __restrict__ out, int n) {
    int stride = gridDim.x * blockDim.x;
    for (int i = blockIdx.x * blockDim.x + threadIdx.x; i < n; i += stride)
        out[i] = f2bf(in[i]);
}

// ---------- WMMA fragment loaders from LDS (ISA 7.12.2 layouts) ----------
// A: 16x32 bf16. lane m = lane&15.
//   lanes 0-15 : elems 0-7 = K[0..7],   elems 8-15 = K[16..23]
//   lanes 16-31: elems 0-7 = K[8..15],  elems 8-15 = K[24..31]
__device__ __forceinline__ v16bf lds_frag_a(const ushort_t* __restrict__ s,
                                            int m0, int lane) {
    const int m   = m0 + (lane & 15);
    const int off = (lane & 16) ? 8 : 0;
    const ushort_t* p = s + m * LDSW + off;
    union { v16bf v; u32x4 u[2]; } f;
    f.u[0] = *(const u32x4*)(p);        // K off .. off+7
    f.u[1] = *(const u32x4*)(p + 16);   // K off+16 .. off+23
    return f.v;
}
// B: 32x16 bf16 (KxN). lane n = lane&15; same K order as A:
//   lanes 0-15: elems 0-7 = K[0..7], 8-15 = K[16..23]; lanes 16-31: elems 0-7 = K[8..15], 8-15 = K[24..31]
__device__ __forceinline__ v16bf lds_frag_b(const ushort_t* __restrict__ s,
                                            int n0, int lane) {
    const int n   = n0 + (lane & 15);
    const int off = (lane & 16) ? 8 : 0;
    const ushort_t* p = s + n * LDSW + off;
    union { v16bf v; u32x4 u[2]; } f;
    f.u[0] = *(const u32x4*)(p);        // K off .. off+7
    f.u[1] = *(const u32x4*)(p + 16);   // K off+16 .. off+23
    return f.v;
}

// ---------- GEMM: out[M,N] = A[M,K] * W[N,K]^T + bias[N] ----------
// Block: 256 threads = 8 waves, 128x128 tile, BK=32, double-buffered LDS.
// Wave grid 2x4, each wave 64x32 (4x2 WMMA tiles).
template <bool OUT_BF16>
__global__ __launch_bounds__(256)
void gemm_bf16_wmma_kernel(const ushort_t* __restrict__ A,
                           const ushort_t* __restrict__ W,
                           const float* __restrict__ bias,
                           void* __restrict__ out,
                           int M, int N, int K) {
    __shared__ __align__(16) ushort_t sA[2][128 * LDSW];
    __shared__ __align__(16) ushort_t sB[2][128 * LDSW];

    const int tid    = threadIdx.x;
    const int lane   = tid & 31;
    const int waveId = tid >> 5;
    const int blockM = blockIdx.y * 128;
    const int blockN = blockIdx.x * 128;
    const int wmL = (waveId >> 2) * 64;   // wave row origin within block
    const int wnL = (waveId & 3) * 32;    // wave col origin within block

    // Cooperative staging: each thread owns one 32B segment of one row
    // (row = tid>>1 in [0,128), seg = (tid&1)*16 elements within the 32-elem row)
    const int srow = tid >> 1;
    const int sseg = (tid & 1) * 16;
    const ushort_t* gA = A + (size_t)(blockM + srow) * K + sseg;
    const ushort_t* gB = W + (size_t)(blockN + srow) * K + sseg;
    const int ldsOff = srow * LDSW + sseg;

#ifdef USE_ASYNC_LDS
    // gfx1250 async DMA path: global -> LDS directly, tracked by ASYNCcnt.
    auto stage_async = [&](int buf, int k0) {
        __builtin_amdgcn_global_load_async_to_lds_b128(
            (gv4i_t)(gA + k0),     (lv4i_t)&sA[buf][ldsOff],     0, 0);
        __builtin_amdgcn_global_load_async_to_lds_b128(
            (gv4i_t)(gA + k0 + 8), (lv4i_t)&sA[buf][ldsOff + 8], 0, 0);
        __builtin_amdgcn_global_load_async_to_lds_b128(
            (gv4i_t)(gB + k0),     (lv4i_t)&sB[buf][ldsOff],     0, 0);
        __builtin_amdgcn_global_load_async_to_lds_b128(
            (gv4i_t)(gB + k0 + 8), (lv4i_t)&sB[buf][ldsOff + 8], 0, 0);
    };
#else
    u32x4 ra0, ra1, rb0, rb1;
    auto load_regs = [&](int k0) {
        ra0 = *(const u32x4*)(gA + k0);
        ra1 = *(const u32x4*)(gA + k0 + 8);
        rb0 = *(const u32x4*)(gB + k0);
        rb1 = *(const u32x4*)(gB + k0 + 8);
    };
    auto store_lds = [&](int buf) {
        *(u32x4*)(&sA[buf][ldsOff])     = ra0;
        *(u32x4*)(&sA[buf][ldsOff + 8]) = ra1;
        *(u32x4*)(&sB[buf][ldsOff])     = rb0;
        *(u32x4*)(&sB[buf][ldsOff + 8]) = rb1;
    };
#endif

    v8f acc[4][2];
#pragma unroll
    for (int i = 0; i < 4; ++i)
#pragma unroll
        for (int j = 0; j < 2; ++j) acc[i][j] = (v8f)0.0f;

    const int nk = K / 32;

#ifdef USE_ASYNC_LDS
    stage_async(0, 0);
    __builtin_amdgcn_s_wait_asynccnt(0);
    __syncthreads();
#else
    load_regs(0);
    store_lds(0);
    __syncthreads();
#endif

    for (int kb = 0; kb < nk; ++kb) {
        const int cur = kb & 1;
        const bool more = (kb + 1) < nk;

#ifdef USE_ASYNC_LDS
        if (more) stage_async(cur ^ 1, (kb + 1) * 32);  // DMA overlaps WMMA
#else
        if (more) load_regs((kb + 1) * 32);             // loads in flight over compute
#endif

        v16bf af[4], bfr[2];
#pragma unroll
        for (int i = 0; i < 4; ++i) af[i]  = lds_frag_a(sA[cur], wmL + i * 16, lane);
#pragma unroll
        for (int j = 0; j < 2; ++j) bfr[j] = lds_frag_b(sB[cur], wnL + j * 16, lane);
#pragma unroll
        for (int i = 0; i < 4; ++i)
#pragma unroll
            for (int j = 0; j < 2; ++j)
                acc[i][j] = __builtin_amdgcn_wmma_f32_16x16x32_bf16(
                    false, af[i], false, bfr[j], (short)0, acc[i][j], false, false);

#ifdef USE_ASYNC_LDS
        if (more) __builtin_amdgcn_s_wait_asynccnt(0);
#else
        if (more) store_lds(cur ^ 1);
#endif
        __syncthreads();
    }

    // Epilogue: C/D layout — lane n = lane&15; VGPR r -> row r + (lane>=16 ? 8 : 0)
    const int mLaneOff = (lane & 16) ? 8 : 0;
    const int nLane    = lane & 15;
#pragma unroll
    for (int j = 0; j < 2; ++j) {
        const int n  = blockN + wnL + j * 16 + nLane;
        const float bv = bias[n];
#pragma unroll
        for (int i = 0; i < 4; ++i) {
#pragma unroll
            for (int r = 0; r < 8; ++r) {
                const int m = blockM + wmL + i * 16 + mLaneOff + r;
                const float val = acc[i][j][r] + bv;
                if (OUT_BF16)
                    ((ushort_t*)out)[(size_t)m * N + n] = f2bf(val);
                else
                    ((float*)out)[(size_t)m * N + n] = val;
            }
        }
    }
}

// ---------- Local windowed attention: one wave (32 lanes) per token ----------
// W = 17 (half=8). Lane covers channel pairs {2*lane + 64*i, +1}, i in [0,16).
__global__ __launch_bounds__(256)
void local_attn_kernel(const ushort_t* __restrict__ q,
                       const ushort_t* __restrict__ k,
                       const ushort_t* __restrict__ v,
                       ushort_t* __restrict__ ctx,
                       int Bn, int Tn, int half) {
    const int C = 1024;
    const int lane   = threadIdx.x & 31;
    const int waveId = threadIdx.x >> 5;
    const int bt = blockIdx.x * 8 + waveId;   // token index in [0, B*T)
    if (bt >= Bn * Tn) return;
    const int b = bt / Tn;
    const int t = bt % Tn;
    const float scale = 0.03125f;             // 1/sqrt(1024)

    // Load q row into registers
    float qr[32];
    const uint_t* qrow = (const uint_t*)(q + (size_t)bt * C);
#pragma unroll
    for (int i = 0; i < 16; ++i) {
        uint_t u = qrow[lane + 32 * i];
        qr[2 * i]     = bf2f((ushort_t)(u & 0xFFFFu));
        qr[2 * i + 1] = bf2f((ushort_t)(u >> 16));
    }

    // Scores over window
    float sc[17];
#pragma unroll 1
    for (int w = 0; w < 17; ++w) {
        const int j = t - half + w;
        float s = -3.0e38f;
        if (j >= 0 && j < Tn) {
            const uint_t* krow = (const uint_t*)(k + ((size_t)b * Tn + j) * C);
            float p = 0.0f;
#pragma unroll
            for (int i = 0; i < 16; ++i) {
                uint_t u = krow[lane + 32 * i];
                p += qr[2 * i]     * bf2f((ushort_t)(u & 0xFFFFu));
                p += qr[2 * i + 1] * bf2f((ushort_t)(u >> 16));
            }
#pragma unroll
            for (int off = 16; off > 0; off >>= 1)
                p += __shfl_xor(p, off, 32);
            s = p * scale;
        }
        sc[w] = s;
    }

    // Softmax (replicated across lanes)
    float mx = sc[0];
#pragma unroll
    for (int w = 1; w < 17; ++w) mx = fmaxf(mx, sc[w]);
    float sum = 0.0f;
#pragma unroll
    for (int w = 0; w < 17; ++w) {
        float e = (sc[w] > -1.0e38f) ? __expf(sc[w] - mx) : 0.0f;
        sc[w] = e;
        sum += e;
    }
    const float inv = 1.0f / sum;

    // Context accumulation
    float cr[32];
#pragma unroll
    for (int i = 0; i < 32; ++i) cr[i] = 0.0f;
#pragma unroll 1
    for (int w = 0; w < 17; ++w) {
        const int j = t - half + w;
        if (j < 0 || j >= Tn) continue;
        const float a = sc[w] * inv;
        const uint_t* vrow = (const uint_t*)(v + ((size_t)b * Tn + j) * C);
#pragma unroll
        for (int i = 0; i < 16; ++i) {
            uint_t u = vrow[lane + 32 * i];
            cr[2 * i]     += a * bf2f((ushort_t)(u & 0xFFFFu));
            cr[2 * i + 1] += a * bf2f((ushort_t)(u >> 16));
        }
    }

    // Store ctx as bf16 (packed dword stores, coalesced)
    uint_t* crow = (uint_t*)(ctx + (size_t)bt * C);
#pragma unroll
    for (int i = 0; i < 16; ++i) {
        uint_t u = (uint_t)f2bf(cr[2 * i]) | ((uint_t)f2bf(cr[2 * i + 1]) << 16);
        crow[lane + 32 * i] = u;
    }
}

extern "C" void kernel_launch(void* const* d_in, const int* in_sizes, int n_in,
                              void* d_out, int out_size, void* d_ws, size_t ws_size,
                              hipStream_t stream) {
    (void)in_sizes; (void)n_in; (void)out_size; (void)ws_size;
    const int B = 4, T = 2048, C = 1024, HALF = 8;
    const int BT = B * T;                     // 8192

    const float* x  = (const float*)d_in[0];
    const float* Wq = (const float*)d_in[1];
    const float* bq = (const float*)d_in[2];
    const float* Wk = (const float*)d_in[3];
    const float* bk = (const float*)d_in[4];
    const float* Wv = (const float*)d_in[5];
    const float* bv = (const float*)d_in[6];
    const float* Wo = (const float*)d_in[7];
    const float* bo = (const float*)d_in[8];
    float* out = (float*)d_out;

    // Workspace partition (bf16 buffers)
    ushort_t* xbf  = (ushort_t*)d_ws;                 // BT*C
    ushort_t* Wqb  = xbf  + (size_t)BT * C;           // C*C
    ushort_t* Wkb  = Wqb  + (size_t)C * C;
    ushort_t* Wvb  = Wkb  + (size_t)C * C;
    ushort_t* Wob  = Wvb  + (size_t)C * C;
    ushort_t* qb   = Wob  + (size_t)C * C;            // BT*C
    ushort_t* kb   = qb   + (size_t)BT * C;
    ushort_t* vb   = kb   + (size_t)BT * C;
    ushort_t* ctxb = vb   + (size_t)BT * C;

    // 1) fp32 -> bf16 conversions
    cvt_f32_bf16_kernel<<<2048, 256, 0, stream>>>(x,  xbf, BT * C);
    cvt_f32_bf16_kernel<<<1024, 256, 0, stream>>>(Wq, Wqb, C * C);
    cvt_f32_bf16_kernel<<<1024, 256, 0, stream>>>(Wk, Wkb, C * C);
    cvt_f32_bf16_kernel<<<1024, 256, 0, stream>>>(Wv, Wvb, C * C);
    cvt_f32_bf16_kernel<<<1024, 256, 0, stream>>>(Wo, Wob, C * C);

    // 2) Q/K/V projections (WMMA bf16, fp32 accumulate), bf16 outputs
    dim3 gGrid(C / 128, BT / 128);                    // (8, 64)
    gemm_bf16_wmma_kernel<true><<<gGrid, 256, 0, stream>>>(xbf, Wqb, bq, qb, BT, C, C);
    gemm_bf16_wmma_kernel<true><<<gGrid, 256, 0, stream>>>(xbf, Wkb, bk, kb, BT, C, C);
    gemm_bf16_wmma_kernel<true><<<gGrid, 256, 0, stream>>>(xbf, Wvb, bv, vb, BT, C, C);

    // 3) Windowed softmax attention -> ctx (bf16)
    local_attn_kernel<<<BT / 8, 256, 0, stream>>>(qb, kb, vb, ctxb, B, T, HALF);

    // 4) Output projection (WMMA), fp32 output
    gemm_bf16_wmma_kernel<false><<<gGrid, 256, 0, stream>>>(ctxb, Wob, bo, out, BT, C, C);
}
